// Mamba_90151363543402
// MI455X (gfx1250) — compile-verified, matrix operand order checked by reading, not run on hardware
//
#include <hip/hip_runtime.h>
#include <cstdint>
#include <cstddef>

// ---------------- model constants ----------------
#define L_SEQ   2048
#define DMODEL  1024
#define DINNER  2048
#define DSTATE  16
#define DCONV   4
#define DTRANK  64
#define NLAYER  4
#define VOCAB   50257
#define VPAD    50304   // vocab padded up to multiple of 128 for GEMM tiles
#define XPROJ_N 96      // DT_RANK + 2*D_STATE
#define XPROJ_NPAD 128

typedef __attribute__((ext_vector_type(16))) __bf16 v16bf;
typedef __attribute__((ext_vector_type(8)))  float  v8f;

// ---------------- helpers ----------------
__device__ __forceinline__ unsigned short f2bf(float f) {
  union { float f; unsigned int u; } c; c.f = f;
  unsigned int u = c.u;
  unsigned int r = u + 0x7FFFu + ((u >> 16) & 1u);   // round-to-nearest-even
  return (unsigned short)(r >> 16);
}
__device__ __forceinline__ float siluf(float x) { return x / (1.f + __expf(-x)); }

union FragBF { v16bf v; uint4 q[2]; };

// ---------------- WMMA bf16 GEMM: C[m,n] = sum_k A[m,k] * W[n,k] ----------------
// A: [M, lda] bf16 row-major, W: [Npad, ldb] bf16 row-major (rows >= Nstore are zero)
// C: [M, ldc] f32. grid = (Npad/128, M/128), block = 256 (8 waves).
// Wave tile 32(M) x 64(N): 2x4 wmma accumulators; K stepped by 32.
// Fragment layout per CDNA5 ISA 7.12.2: lane row = lane&15,
// K halves [klo..klo+7] and [klo+16..klo+23] with klo = (lane>>4)*8.
__global__ void __launch_bounds__(256)
k_gemm_bf16(const unsigned short* __restrict__ A, int lda,
            const unsigned short* __restrict__ B, int ldb,
            float* __restrict__ C, int ldc,
            int K, int Nstore)
{
  const int lane = threadIdx.x & 31;
  const int wave = threadIdx.x >> 5;
  const int wm   = wave & 3;                 // 4 M-subtiles of 32
  const int wn   = wave >> 2;                // 2 N-subtiles of 64
  const int m_wave = blockIdx.y * 128 + wm * 32;
  const int n_wave = blockIdx.x * 128 + wn * 64;

  const int row = lane & 15;
  const int klo = (lane >> 4) << 3;          // 0 or 8

  v8f acc[2][4];
#pragma unroll
  for (int mi = 0; mi < 2; ++mi)
#pragma unroll
    for (int ni = 0; ni < 4; ++ni)
#pragma unroll
      for (int r = 0; r < 8; ++r) acc[mi][ni][r] = 0.f;

  for (int k0 = 0; k0 < K; k0 += 32) {
    FragBF a[2], b[4];
#pragma unroll
    for (int mi = 0; mi < 2; ++mi) {
      const unsigned short* p = A + (size_t)(m_wave + mi * 16 + row) * lda + k0 + klo;
      a[mi].q[0] = *(const uint4*)(p);
      a[mi].q[1] = *(const uint4*)(p + 16);
    }
#pragma unroll
    for (int ni = 0; ni < 4; ++ni) {
      const unsigned short* p = B + (size_t)(n_wave + ni * 16 + row) * ldb + k0 + klo;
      b[ni].q[0] = *(const uint4*)(p);
      b[ni].q[1] = *(const uint4*)(p + 16);
    }
#pragma unroll
    for (int mi = 0; mi < 2; ++mi)
#pragma unroll
      for (int ni = 0; ni < 4; ++ni)
        acc[mi][ni] = __builtin_amdgcn_wmma_f32_16x16x32_bf16(
            false, a[mi].v, false, b[ni].v, (short)0, acc[mi][ni], false, false);
  }

  // C/D layout (7.12.2): lane col = lane&15; element r -> m = (lane>=16?8:0)+r
  const int colC   = lane & 15;
  const int rowOff = (lane >> 4) * 8;
#pragma unroll
  for (int mi = 0; mi < 2; ++mi)
#pragma unroll
    for (int ni = 0; ni < 4; ++ni) {
      const int n = n_wave + ni * 16 + colC;
      if (n < Nstore) {
#pragma unroll
        for (int r = 0; r < 8; ++r) {
          const int m = m_wave + mi * 16 + rowOff + r;
          C[(size_t)m * ldc + n] = acc[mi][ni][r];
        }
      }
    }
}

// ---------------- weight f32 -> bf16 with zero row padding ----------------
__global__ void __launch_bounds__(256)
k_convert_w(const float* __restrict__ W, unsigned short* __restrict__ out,
            int Nlog, int K)
{
  const int idx = blockIdx.x * 256 + threadIdx.x;      // over Npad*K
  const int n = idx / K;
  out[idx] = (n < Nlog) ? f2bf(W[idx]) : (unsigned short)0;
}

// ---------------- embedding gather ----------------
__global__ void __launch_bounds__(256)
k_gather(const int* __restrict__ ids, const float* __restrict__ embed,
         float* __restrict__ x)
{
  const int idx = blockIdx.x * 256 + threadIdx.x;      // over L*DMODEL
  const int l = idx / DMODEL, d = idx % DMODEL;
  x[idx] = embed[(size_t)ids[l] * DMODEL + d];
}

// ---------------- RMSNorm -> bf16 ----------------
__global__ void __launch_bounds__(256)
k_rmsnorm_bf16(const float* __restrict__ x, const float* __restrict__ w,
               unsigned short* __restrict__ out)
{
  __shared__ float red[256];
  const int l = blockIdx.x;
  const int t = threadIdx.x;
  const float* xr = x + (size_t)l * DMODEL;
  float v[4]; float s = 0.f;
#pragma unroll
  for (int j = 0; j < 4; ++j) { v[j] = xr[t + j * 256]; s += v[j] * v[j]; }
  red[t] = s; __syncthreads();
  for (int o = 128; o > 0; o >>= 1) { if (t < o) red[t] += red[t + o]; __syncthreads(); }
  const float inv = rsqrtf(red[0] / (float)DMODEL + 1e-5f);
  unsigned short* orow = out + (size_t)l * DMODEL;
#pragma unroll
  for (int j = 0; j < 4; ++j) orow[t + j * 256] = f2bf(v[j] * inv * w[t + j * 256]);
}

// ---------------- depthwise causal conv (k=4) + SiLU ----------------
__global__ void __launch_bounds__(256)
k_conv_silu(const float* __restrict__ xz, const float* __restrict__ cw,
            const float* __restrict__ cb, float* __restrict__ xi_f,
            unsigned short* __restrict__ xi_bf)
{
  const int idx = blockIdx.x * 256 + threadIdx.x;      // over L*DINNER
  const int l = idx / DINNER, d = idx % DINNER;
  float acc = cb[d];
#pragma unroll
  for (int j = 0; j < DCONV; ++j) {
    const int t = l - (DCONV - 1) + j;
    if (t >= 0) acc += cw[d * DCONV + j] * xz[(size_t)t * (2 * DINNER) + d];
  }
  const float s = siluf(acc);
  xi_f[idx] = s;
  xi_bf[idx] = f2bf(s);
}

// ---------------- extract dt columns of xdbc -> bf16 ----------------
__global__ void __launch_bounds__(256)
k_extract_dt(const float* __restrict__ xdbc, unsigned short* __restrict__ dtin)
{
  const int idx = blockIdx.x * 256 + threadIdx.x;      // over L*DTRANK
  const int l = idx / DTRANK, r = idx % DTRANK;
  dtin[idx] = f2bf(xdbc[(size_t)l * XPROJ_NPAD + r]);
}

// ---------------- dt = softplus(dt_raw + bias), in place ----------------
__global__ void __launch_bounds__(256)
k_dt_act(float* __restrict__ dt, const float* __restrict__ bias)
{
  const int idx = blockIdx.x * 256 + threadIdx.x;      // over L*DINNER
  const int d = idx % DINNER;
  const float v = dt[idx] + bias[d];
  dt[idx] = (v > 20.f) ? v : log1pf(__expf(v));
}

// ---------------- selective scan: 1 lane per channel, 16 states in VGPRs ----------------
__global__ void __launch_bounds__(256)
k_scan(const float* __restrict__ dt, const float* __restrict__ xi,
       const float* __restrict__ xdbc, const float* __restrict__ A_log,
       float* __restrict__ y)
{
  const int d = blockIdx.x * 256 + threadIdx.x;        // < DINNER
  float A[DSTATE], h[DSTATE];
#pragma unroll
  for (int s = 0; s < DSTATE; ++s) {
    A[s] = -__expf(A_log[(size_t)d * DSTATE + s]);
    h[s] = 0.f;
  }
  for (int t = 0; t < L_SEQ; ++t) {
    const float dtv = dt[(size_t)t * DINNER + d];
    const float dx  = dtv * xi[(size_t)t * DINNER + d];
    const float* bc = xdbc + (size_t)t * XPROJ_NPAD + DTRANK;   // B then C
    float ys = 0.f;
#pragma unroll
    for (int s = 0; s < DSTATE; ++s) {
      const float dA = __expf(dtv * A[s]);
      h[s] = dA * h[s] + dx * bc[s];
      ys += h[s] * bc[DSTATE + s];
    }
    y[(size_t)t * DINNER + d] = ys;
  }
}

// ---------------- y = (y + D*xi) * silu(z) -> bf16 ----------------
__global__ void __launch_bounds__(256)
k_gate(const float* __restrict__ y, const float* __restrict__ xi_f,
       const float* __restrict__ Dp, const float* __restrict__ xz,
       unsigned short* __restrict__ ybf)
{
  const int idx = blockIdx.x * 256 + threadIdx.x;      // over L*DINNER
  const int l = idx / DINNER, d = idx % DINNER;
  const float z = xz[(size_t)l * (2 * DINNER) + DINNER + d];
  ybf[idx] = f2bf((y[idx] + Dp[d] * xi_f[idx]) * siluf(z));
}

// ---------------- residual add ----------------
__global__ void __launch_bounds__(256)
k_add(float* __restrict__ x, const float* __restrict__ t)
{
  const int idx = blockIdx.x * 256 + threadIdx.x;      // over L*DMODEL
  x[idx] += t[idx];
}

// ---------------- host orchestration ----------------
// Input order (jax tree leaves, dicts sorted by key):
//  0 input_ids[i32 2048], 1 embed[50257*1024],
//  2 A_log[4,2048,16], 3 D[4,2048], 4 conv_b[4,2048], 5 conv_w[4,2048,1,4],
//  6 dt_bias[4,2048], 7 dt_proj[4,2048,64], 8 in_proj[4,4096,1024],
//  9 norm[4,1024], 10 out_proj[4,1024,2048], 11 x_proj[4,96,2048],
//  12 norm_f[1024]
extern "C" void kernel_launch(void* const* d_in, const int* in_sizes, int n_in,
                              void* d_out, int out_size, void* d_ws, size_t ws_size,
                              hipStream_t stream) {
  (void)in_sizes; (void)n_in; (void)out_size; (void)ws_size;
  const int*   ids       = (const int*)  d_in[0];
  const float* embed     = (const float*)d_in[1];
  const float* A_log     = (const float*)d_in[2];
  const float* Dp        = (const float*)d_in[3];
  const float* conv_b    = (const float*)d_in[4];
  const float* conv_w    = (const float*)d_in[5];
  const float* dt_bias   = (const float*)d_in[6];
  const float* dt_projW  = (const float*)d_in[7];
  const float* in_projW  = (const float*)d_in[8];
  const float* norm_w    = (const float*)d_in[9];
  const float* out_projW = (const float*)d_in[10];
  const float* x_projW   = (const float*)d_in[11];
  const float* norm_f    = (const float*)d_in[12];
  float* logits = (float*)d_out;

  char* ws = (char*)d_ws; size_t off = 0;
  auto alloc = [&](size_t bytes) -> void* {
    void* p = ws + off; off += (bytes + 255) & ~(size_t)255; return p;
  };
  float*          xres  = (float*)         alloc((size_t)L_SEQ * DMODEL * 4);
  unsigned short* xnbf  = (unsigned short*)alloc((size_t)L_SEQ * DMODEL * 2);
  unsigned short* wbuf  = (unsigned short*)alloc((size_t)VPAD * DMODEL * 2);
  float*          xz    = (float*)         alloc((size_t)L_SEQ * 2 * DINNER * 4);
  float*          xi_f  = (float*)         alloc((size_t)L_SEQ * DINNER * 4);
  unsigned short* xi_bf = (unsigned short*)alloc((size_t)L_SEQ * DINNER * 2);
  float*          xdbc  = (float*)         alloc((size_t)L_SEQ * XPROJ_NPAD * 4);
  unsigned short* dtin  = (unsigned short*)alloc((size_t)L_SEQ * DTRANK * 2);
  float*          dtbuf = (float*)         alloc((size_t)L_SEQ * DINNER * 4);
  float*          ybuf  = (float*)         alloc((size_t)L_SEQ * DINNER * 4);
  unsigned short* ybf   = (unsigned short*)alloc((size_t)L_SEQ * DINNER * 2);
  float*          projo = (float*)         alloc((size_t)L_SEQ * DMODEL * 4);

  const dim3 blk(256);
  const int gLD  = (L_SEQ * DMODEL) / 256;
  const int gLDI = (L_SEQ * DINNER) / 256;

  k_gather<<<gLD, blk, 0, stream>>>(ids, embed, xres);

  for (int li = 0; li < NLAYER; ++li) {
    const float* w_norm = norm_w    + (size_t)li * DMODEL;
    const float* w_in   = in_projW  + (size_t)li * 2 * DINNER * DMODEL;
    const float* w_x    = x_projW   + (size_t)li * XPROJ_N * DINNER;
    const float* w_dt   = dt_projW  + (size_t)li * DINNER * DTRANK;
    const float* w_out  = out_projW + (size_t)li * DMODEL * DINNER;
    const float* l_cw   = conv_w    + (size_t)li * DINNER * DCONV;
    const float* l_cb   = conv_b    + (size_t)li * DINNER;
    const float* l_db   = dt_bias   + (size_t)li * DINNER;
    const float* l_Al   = A_log     + (size_t)li * DINNER * DSTATE;
    const float* l_D    = Dp        + (size_t)li * DINNER;

    // xn = rmsnorm(x) -> bf16
    k_rmsnorm_bf16<<<L_SEQ, blk, 0, stream>>>(xres, w_norm, xnbf);

    // xz = xn @ in_proj^T   [2048 x 4096]
    k_convert_w<<<(2 * DINNER * DMODEL) / 256, blk, 0, stream>>>(w_in, wbuf, 2 * DINNER, DMODEL);
    k_gemm_bf16<<<dim3(2 * DINNER / 128, L_SEQ / 128), blk, 0, stream>>>(
        xnbf, DMODEL, wbuf, DMODEL, xz, 2 * DINNER, DMODEL, 2 * DINNER);

    // xi = silu(depthwise_conv(xz[:, :DINNER]))
    k_conv_silu<<<gLDI, blk, 0, stream>>>(xz, l_cw, l_cb, xi_f, xi_bf);

    // xdbc = xi @ x_proj^T   [2048 x 96] (padded to 128)
    k_convert_w<<<(XPROJ_NPAD * DINNER) / 256, blk, 0, stream>>>(w_x, wbuf, XPROJ_N, DINNER);
    k_gemm_bf16<<<dim3(XPROJ_NPAD / 128, L_SEQ / 128), blk, 0, stream>>>(
        xi_bf, DINNER, wbuf, DINNER, xdbc, XPROJ_NPAD, DINNER, XPROJ_N);

    // dt_raw = xdbc[:, :64] @ dt_proj^T   [2048 x 2048]
    k_extract_dt<<<(L_SEQ * DTRANK) / 256, blk, 0, stream>>>(xdbc, dtin);
    k_convert_w<<<(DINNER * DTRANK) / 256, blk, 0, stream>>>(w_dt, wbuf, DINNER, DTRANK);
    k_gemm_bf16<<<dim3(DINNER / 128, L_SEQ / 128), blk, 0, stream>>>(
        dtin, DTRANK, wbuf, DTRANK, dtbuf, DINNER, DTRANK, DINNER);
    k_dt_act<<<gLDI, blk, 0, stream>>>(dtbuf, l_db);

    // selective scan
    k_scan<<<DINNER / 256, blk, 0, stream>>>(dtbuf, xi_f, xdbc, l_Al, ybuf);

    // gate: y = (y + D*xi) * silu(z)
    k_gate<<<gLDI, blk, 0, stream>>>(ybuf, xi_f, l_D, xz, ybf);

    // proj = y @ out_proj^T   [2048 x 1024]; x += proj
    k_convert_w<<<(DMODEL * DINNER) / 256, blk, 0, stream>>>(w_out, wbuf, DMODEL, DINNER);
    k_gemm_bf16<<<dim3(DMODEL / 128, L_SEQ / 128), blk, 0, stream>>>(
        ybf, DINNER, wbuf, DINNER, projo, DMODEL, DINNER, DMODEL);
    k_add<<<gLD, blk, 0, stream>>>(xres, projo);
  }

  // final norm + tied lm_head
  k_rmsnorm_bf16<<<L_SEQ, blk, 0, stream>>>(xres, norm_f, xnbf);
  k_convert_w<<<((size_t)VPAD * DMODEL) / 256, blk, 0, stream>>>(embed, wbuf, VOCAB, DMODEL);
  k_gemm_bf16<<<dim3(VPAD / 128, L_SEQ / 128), blk, 0, stream>>>(
      xnbf, DMODEL, wbuf, DMODEL, logits, VOCAB, DMODEL, VOCAB);
}
